// GRU_42640435315362
// MI455X (gfx1250) — matrix operand order repaired and checked by reading, not compiled, not run on hardware
//
#include <hip/hip_runtime.h>

typedef __attribute__((ext_vector_type(16))) __bf16        v16bf;
typedef __attribute__((ext_vector_type(8)))  float         v8f;
typedef __attribute__((ext_vector_type(4)))  unsigned int  uint4v;

constexpr int Dm   = 128;        // hidden / emb width
constexpr int TT   = 20;         // sequence length
constexpr int BSz  = 1024;       // batch
constexpr int Mrow = BSz * TT;   // 20480 rows for gx GEMM
constexpr int KCAT = 160;        // 144 concat width padded to 5*32
constexpr int G3   = 384;        // 3*H gates
constexpr int HBS  = 136;        // padded LDS row stride for h tile (272B = 68 dwords,
                                 // 68 mod 64 = 4 -> each row shifted 4 banks: conflict-free)
constexpr float INV_TEMP = 20.0f; // 1/0.05

__device__ __forceinline__ unsigned short f2bf(float f) {
  unsigned int u = __float_as_uint(f);
  u += 0x7FFFu + ((u >> 16) & 1u);   // round-to-nearest-even
  return (unsigned short)(u >> 16);
}

union FragBF { v16bf v; uint4v q[2]; };

// A-matrix fragment (16x32 bf16): row m = lane&15; VGPR0..3 hold K=8h..8h+7,
// VGPR4..7 hold K=16+8h..16+8h+7  -> two contiguous 16B loads per lane.
__device__ __forceinline__ v16bf load_a_frag(const unsigned short* A, int lda, int k0) {
  int l = threadIdx.x & 31;
  int hl = l >> 4, m = l & 15;
  FragBF f;
  const unsigned short* p = A + m * lda + k0 + 8 * hl;
  f.q[0] = *(const uint4v*)(p);
  f.q[1] = *(const uint4v*)(p + 16);
  return f.v;
}

// B-matrix fragment (32x16 bf16), operand stored as Bt[N][K]: col n = lane&15,
// VGPR0..3 hold K=8h..8h+7, VGPR4..7 hold K=16+8h..16+8h+7 -> two 16B loads per lane.
__device__ __forceinline__ v16bf load_b_frag(const unsigned short* Bt, int ldb, int k0) {
  int l = threadIdx.x & 31;
  int hl = l >> 4, n = l & 15;
  FragBF f;
  const unsigned short* p = Bt + n * ldb + k0 + 8 * hl;
  f.q[0] = *(const uint4v*)(p);
  f.q[1] = *(const uint4v*)(p + 16);
  return f.v;
}

__device__ __forceinline__ v8f wmma_bf16(v16bf a, v16bf b, v8f c) {
  return __builtin_amdgcn_wmma_f32_16x16x32_bf16(false, a, false, b, (short)0, c, false, false);
}

// ---------------- prep: W_eff = W_mlp @ W_ih^T (144x384, pad K->160),
//                        b_eff = b_ih + b_mlp @ W_ih^T, W_hh -> bf16 [384][128]
__global__ __launch_bounds__(256) void prep_kernel(
    const float* __restrict__ W_mlp, const float* __restrict__ b_mlp,
    const float* __restrict__ W_ih,  const float* __restrict__ W_hh,
    const float* __restrict__ b_ih,
    unsigned short* __restrict__ W_eff_t, float* __restrict__ b_eff,
    unsigned short* __restrict__ W_hh_t) {
  int idx = blockIdx.x * 256 + threadIdx.x;
  if (idx < G3 * KCAT) {
    int g = idx / KCAT, k = idx % KCAT;
    float s = 0.f;
    if (k < 144) {
      for (int c = 0; c < Dm; ++c) s += W_mlp[k * Dm + c] * W_ih[g * Dm + c];
    }
    W_eff_t[g * KCAT + k] = f2bf(s);
  } else if (idx < G3 * KCAT + G3) {
    int g = idx - G3 * KCAT;
    float s = b_ih[g];
    for (int c = 0; c < Dm; ++c) s += b_mlp[c] * W_ih[g * Dm + c];
    b_eff[g] = s;
  } else if (idx < G3 * KCAT + G3 + G3 * Dm) {
    int i = idx - (G3 * KCAT + G3);
    W_hh_t[i] = f2bf(W_hh[i]);
  }
}

// ---------------- gather + concat embeddings into bf16 A matrix [20480][160]
__global__ __launch_bounds__(256) void build_a_kernel(
    const int* __restrict__ aid, const int* __restrict__ typ, const int* __restrict__ rnk,
    const float* __restrict__ emb_item, const float* __restrict__ emb_type,
    const float* __restrict__ emb_rank, unsigned short* __restrict__ Acat) {
  int idx = blockIdx.x * 256 + threadIdx.x;
  if (idx >= Mrow * KCAT) return;
  int m = idx / KCAT, k = idx % KCAT;
  float v = 0.f;
  if (k < Dm)       v = emb_item[(long)aid[m] * Dm + k];
  else if (k < 136) v = emb_type[typ[m] * 8 + (k - 128)];
  else if (k < 144) v = emb_rank[rnk[m] * 8 + (k - 136)];
  Acat[idx] = f2bf(v);
}

// ---------------- gx = Acat @ W_eff + b_eff  : M=20480, K=160, N=384, f32 out
__global__ __launch_bounds__(256) void gx_gemm_kernel(
    const unsigned short* __restrict__ Acat, const unsigned short* __restrict__ W_eff_t,
    const float* __restrict__ b_eff, float* __restrict__ gx) {
  int wid = blockIdx.x * 8 + (threadIdx.x >> 5);
  int mt = wid / (G3 / 16);
  int nt = wid % (G3 / 16);
  const unsigned short* A = Acat + (long)mt * 16 * KCAT;
  const unsigned short* B = W_eff_t + nt * 16 * KCAT;
  v8f c = {};
#pragma unroll
  for (int k0 = 0; k0 < KCAT; k0 += 32)
    c = wmma_bf16(load_a_frag(A, KCAT, k0), load_b_frag(B, KCAT, k0), c);
  int l = threadIdx.x & 31, hl = l >> 4, n = l & 15;
  int col = nt * 16 + n;
  float be = b_eff[col];
#pragma unroll
  for (int v = 0; v < 8; ++v) {
    int m = mt * 16 + v + 8 * hl;
    gx[(long)m * G3 + col] = c[v] + be;
  }
}

// ---------------- GRU recurrence: 16 batch rows per block, h resident in LDS.
// 8 waves; wave w owns hidden cols [16w,16w+16). W_hh fragments are hoisted into
// registers (12 frags = 96 VGPRs/lane) so the serial loop issues no weight loads.
__global__ __launch_bounds__(256) void gru_kernel(
    const float* __restrict__ gx, const unsigned short* __restrict__ W_hh_t,
    const float* __restrict__ b_hh, const int* __restrict__ sess_len,
    unsigned short* __restrict__ xlast) {
  __shared__ __align__(16) float          h32[16 * Dm];
  __shared__ __align__(16) unsigned short hbf[16 * HBS];
  int tid = threadIdx.x;
  int wave = tid >> 5, l = tid & 31, hl = l >> 4, n = l & 15;
  int col = wave * 16 + n;
  int b0 = blockIdx.x * 16;
  for (int i = tid; i < 16 * Dm; i += 256) h32[i] = 0.f;
  for (int i = tid; i < 16 * HBS; i += 256) hbf[i] = 0;
  float bhr = b_hh[col], bhz = b_hh[col + Dm], bhn = b_hh[col + 2 * Dm];
  int sl[8];
#pragma unroll
  for (int v = 0; v < 8; ++v) sl[v] = sess_len[b0 + v + 8 * hl];
  const unsigned short* Br = W_hh_t + (0 * Dm + wave * 16) * Dm;
  const unsigned short* Bz = W_hh_t + (1 * Dm + wave * 16) * Dm;
  const unsigned short* Bn = W_hh_t + (2 * Dm + wave * 16) * Dm;
  v16bf br[4], bz[4], bn[4];
#pragma unroll
  for (int i = 0; i < 4; ++i) {
    br[i] = load_b_frag(Br, Dm, 32 * i);
    bz[i] = load_b_frag(Bz, Dm, 32 * i);
    bn[i] = load_b_frag(Bn, Dm, 32 * i);
  }
  __syncthreads();
  for (int t = 0; t < TT; ++t) {
    v16bf a[4];
#pragma unroll
    for (int i = 0; i < 4; ++i) a[i] = load_a_frag(hbf, HBS, 32 * i);
    __syncthreads();                      // all reads of hbf done before rewrites
    if (t + 1 < TT) {                     // warm L2/L0 for next step's gx rows
#pragma unroll
      for (int v = 0; v < 8; v += 4) {
        long gxp = ((long)(b0 + v + 8 * hl) * TT + (t + 1)) * G3;
        __builtin_prefetch(&gx[gxp + col], 0, 1);
        __builtin_prefetch(&gx[gxp + Dm + col], 0, 1);
        __builtin_prefetch(&gx[gxp + 2 * Dm + col], 0, 1);
      }
    }
    v8f cr = {}, cz = {}, cn = {};
#pragma unroll
    for (int i = 0; i < 4; ++i) {
      cr = wmma_bf16(a[i], br[i], cr);
      cz = wmma_bf16(a[i], bz[i], cz);
      cn = wmma_bf16(a[i], bn[i], cn);
    }
#pragma unroll
    for (int v = 0; v < 8; ++v) {
      int rl = v + 8 * hl;
      int b  = b0 + rl;
      long gxb = ((long)b * TT + t) * G3;
      float gr = gx[gxb + col];
      float gz = gx[gxb + Dm + col];
      float gn = gx[gxb + 2 * Dm + col];
      float hold = h32[rl * Dm + col];
      float r  = 1.f / (1.f + __expf(-(gr + cr[v] + bhr)));
      float z  = 1.f / (1.f + __expf(-(gz + cz[v] + bhz)));
      float nn = tanhf(gn + r * (cn[v] + bhn));
      float hnew = (1.f - z) * nn + z * hold;
      h32[rl * Dm + col] = hnew;
      hbf[rl * HBS + col] = f2bf(hnew);
      if (t == sl[v] - 1) xlast[(long)b * Dm + col] = f2bf(hnew);
    }
    __syncthreads();                      // new h visible before next step's reads
  }
}

// ---------------- candidate embedding gather -> bf16 [Upad][128]
__global__ __launch_bounds__(256) void cand_kernel(
    const int* __restrict__ uniq, const float* __restrict__ emb_item,
    unsigned short* __restrict__ cand, int U, int Upad) {
  int idx = blockIdx.x * 256 + threadIdx.x;
  if (idx >= Upad * Dm) return;
  int j = idx / Dm, k = idx % Dm;
  float v = 0.f;
  if (j < U) v = emb_item[(long)uniq[j] * Dm + k];
  cand[idx] = f2bf(v);
}

// ---------------- pos_idx = searchsorted(unique_labels, labels)
__global__ __launch_bounds__(256) void pos_kernel(
    const int* __restrict__ uniq, const int* __restrict__ labels,
    int* __restrict__ pos, int U) {
  int b = blockIdx.x * 256 + threadIdx.x;
  if (b >= BSz) return;
  int lab = labels[b], lo = 0, hi = U;
  while (lo < hi) { int mid = (lo + hi) >> 1; if (uniq[mid] < lab) lo = mid + 1; else hi = mid; }
  pos[b] = lo;
}

// ---------------- scores GEMM + history mask + positive-first reorder + 1/temp.
// 4 waves/block share one 16-row M tile (aid history staged in LDS), 4 N tiles.
__global__ __launch_bounds__(128) void score_kernel(
    const unsigned short* __restrict__ xlast, const unsigned short* __restrict__ cand,
    const int* __restrict__ uniq, const int* __restrict__ aid,
    const int* __restrict__ posi, float* __restrict__ out, int U, int Nt4) {
  int wave = threadIdx.x >> 5, l = threadIdx.x & 31, hl = l >> 4, n = l & 15;
  int mt = blockIdx.x / Nt4;
  int nt = (blockIdx.x % Nt4) * 4 + wave;
  __shared__ int aid_s[16 * TT];
  __shared__ int pos_s[16];
  for (int i = threadIdx.x; i < 16 * TT; i += 128) {
    int r = i / TT, t = i % TT;
    aid_s[i] = aid[(mt * 16 + r) * TT + t];
  }
  if (threadIdx.x < 16) pos_s[threadIdx.x] = posi[mt * 16 + threadIdx.x];
  __syncthreads();
  const unsigned short* A = xlast + (long)mt * 16 * Dm;
  const unsigned short* B = cand + (long)nt * 16 * Dm;
  v8f c = {};
#pragma unroll
  for (int k0 = 0; k0 < Dm; k0 += 32)
    c = wmma_bf16(load_a_frag(A, Dm, k0), load_b_frag(B, Dm, k0), c);
  int j = nt * 16 + n;
  if (j >= U) return;                      // only padded tail columns exit
  int u = uniq[j];
#pragma unroll
  for (int v = 0; v < 8; ++v) {
    int rl = v + 8 * hl;
    int b  = mt * 16 + rl;
    bool isin = false;
#pragma unroll
    for (int t = 0; t < TT; ++t) isin |= (aid_s[rl * TT + t] == u);
    int p = pos_s[rl];
    float val = ((!isin) || (j == p)) ? c[v] * INV_TEMP : 0.f;
    int oc = (j == p) ? 0 : (j < p ? j + 1 : j);
    out[(long)b * U + oc] = val;
  }
}

extern "C" void kernel_launch(void* const* d_in, const int* in_sizes, int n_in,
                              void* d_out, int out_size, void* d_ws, size_t ws_size,
                              hipStream_t stream) {
  const int*   aid    = (const int*)d_in[0];
  const int*   typ    = (const int*)d_in[1];
  const int*   rnk    = (const int*)d_in[2];
  const int*   slen   = (const int*)d_in[3];
  const int*   labels = (const int*)d_in[4];
  const int*   uniq   = (const int*)d_in[5];
  const float* emb_item = (const float*)d_in[6];
  const float* emb_type = (const float*)d_in[7];
  const float* emb_rank = (const float*)d_in[8];
  const float* W_mlp = (const float*)d_in[9];
  const float* b_mlp = (const float*)d_in[10];
  const float* W_ih  = (const float*)d_in[11];
  const float* W_hh  = (const float*)d_in[12];
  const float* b_ih  = (const float*)d_in[13];
  const float* b_hh  = (const float*)d_in[14];
  float* out = (float*)d_out;

  int U = in_sizes[5];
  int Upad = ((U + 63) / 64) * 64;

  char* ws = (char*)d_ws;
  size_t off = 0;
  auto carve = [&](size_t bytes) -> char* {
    char* p = ws + off;
    off += (bytes + 255) & ~(size_t)255;
    return p;
  };
  unsigned short* W_eff_t = (unsigned short*)carve((size_t)G3 * KCAT * 2);
  float*          b_eff   = (float*)carve((size_t)G3 * 4);
  unsigned short* W_hh_t  = (unsigned short*)carve((size_t)G3 * Dm * 2);
  unsigned short* Acat    = (unsigned short*)carve((size_t)Mrow * KCAT * 2);
  float*          gx      = (float*)carve((size_t)Mrow * G3 * 4);
  unsigned short* xlast   = (unsigned short*)carve((size_t)BSz * Dm * 2);
  unsigned short* cand    = (unsigned short*)carve((size_t)Upad * Dm * 2);
  int*            posi    = (int*)carve((size_t)BSz * 4);

  {
    int nthr = G3 * KCAT + G3 + G3 * Dm;
    prep_kernel<<<(nthr + 255) / 256, 256, 0, stream>>>(W_mlp, b_mlp, W_ih, W_hh, b_ih,
                                                        W_eff_t, b_eff, W_hh_t);
  }
  {
    int nthr = Mrow * KCAT;
    build_a_kernel<<<(nthr + 255) / 256, 256, 0, stream>>>(aid, typ, rnk, emb_item, emb_type,
                                                           emb_rank, Acat);
  }
  {
    int nblk = (Mrow / 16) * (G3 / 16) / 8;   // 8 waves per block, 1 tile per wave
    gx_gemm_kernel<<<nblk, 256, 0, stream>>>(Acat, W_eff_t, b_eff, gx);
  }
  gru_kernel<<<BSz / 16, 256, 0, stream>>>(gx, W_hh_t, b_hh, slen, xlast);
  {
    int nthr = Upad * Dm;
    cand_kernel<<<(nthr + 255) / 256, 256, 0, stream>>>(uniq, emb_item, cand, U, Upad);
  }
  pos_kernel<<<(BSz + 255) / 256, 256, 0, stream>>>(uniq, labels, posi, U);
  {
    int Nt4 = Upad / 64;
    score_kernel<<<(BSz / 16) * Nt4, 128, 0, stream>>>(xlast, cand, uniq, aid, posi, out, U, Nt4);
  }
}
